// Encoder_50414326120900
// MI455X (gfx1250) — matrix operand order repaired and checked by reading, not compiled, not run on hardware
//
#include <hip/hip_runtime.h>

// ---------------------------------------------------------------------------
// CDNA5 / gfx1250: implicit-GEMM bf16 WMMA convs (v_wmma_f32_16x16x32_bf16),
// weight tiles streamed into LDS by the Tensor Data Mover (tensor_load_to_lds
// + s_wait_tensorcnt), layer dims templated so im2col address math is
// shifts/constant-muls instead of runtime division.
// ---------------------------------------------------------------------------

typedef __bf16 bf16;
typedef __bf16 v16bf  __attribute__((ext_vector_type(16)));
typedef __bf16 bf16x8 __attribute__((ext_vector_type(8)));
typedef float  v8f    __attribute__((ext_vector_type(8)));
typedef unsigned int u32x4 __attribute__((ext_vector_type(4)));
typedef int          i32x4 __attribute__((ext_vector_type(4)));
typedef int          i32x8 __attribute__((ext_vector_type(8)));

// ------------------------------- utilities ---------------------------------

__launch_bounds__(256)
__global__ void zero_f32(float* __restrict__ p, int n) {
    int i = blockIdx.x * 256 + threadIdx.x;
    if (i < n) p[i] = 0.f;
}

__launch_bounds__(256)
__global__ void pack_bf16_k(const float* __restrict__ in, bf16* __restrict__ out, int n) {
    int i = blockIdx.x * 256 + threadIdx.x;
    if (i < n) out[i] = (bf16)in[i];
}

// Pre-pack conv weights: fp32 [Cout][Kreal] -> bf16 [Cout][Kpad] zero-padded.
__launch_bounds__(256)
__global__ void wpack_k(const float* __restrict__ w, bf16* __restrict__ out,
                        int Cout, int Kreal, int Kpad) {
    int i = blockIdx.x * 256 + threadIdx.x;
    if (i < Cout * Kpad) {
        int co = i / Kpad, k = i - co * Kpad;
        out[i] = (bf16)((k < Kreal) ? w[co * Kreal + k] : 0.f);
    }
}

// --------------------------- WMMA implicit GEMM ----------------------------
// GEMM: M = Cout, K = Cin*KS*KS (padded to mult 32), N = 1024*OH*OW.
// WG: 256 threads = 8 waves tiled 2(M) x 4(N); WG tile 32 x 128.
// Weight tile (32x32 bf16) DMA'd into LDS by the TDM each K-step with LDS
// padding 16dw+4dw -> As[32][40] layout; B tile im2col-staged by VALU.

template<int Cin, int CinS, int COFF, int Cout,
         int IH, int IW, int OH, int OW, int STRIDE, int PAD, int KS>
__launch_bounds__(256)
__global__ void conv_wmma(const bf16* __restrict__ in, const bf16* __restrict__ wbf,
                          float* __restrict__ y) {
    constexpr int Kreal  = Cin * KS * KS;
    constexpr int Kpad   = (Kreal + 31) & ~31;
    constexpr int Ksteps = Kpad / 32;
    constexpr int OHOW   = OH * OW;
    constexpr int IHIW   = IH * IW;
    constexpr int Npix   = 1024 * OHOW;

    __shared__ __align__(16) bf16 As[32][40];    // [co][k] (TDM-padded rows)
    __shared__ __align__(16) bf16 Bs[128][40];   // [pixel][k]

    const int tid  = threadIdx.x;
    const int p0   = blockIdx.x * 128;
    const int m0   = blockIdx.y * 32;

    const int lane = tid & 31;
    const int wave = tid >> 5;
    const int wm   = wave >> 2;     // 0..1 : 16-row M sub-tile
    const int wn   = wave & 3;      // 0..3 : 32-col N sub-tile
    const int row  = lane & 15;
    const int kh   = lane >> 4;     // K-half selector (this code's contiguous k order)

    const unsigned lds_as = (unsigned)(unsigned long long)(void*)&As[0][0];

    v8f c0 = {};
    v8f c1 = {};

    for (int ks = 0; ks < Ksteps; ++ks) {
        const int k0 = ks << 5;

        // ---- TDM: async 32x32 bf16 weight tile -> LDS (wave 0 issues) ----
        {
            unsigned long long ga =
                (unsigned long long)(const void*)(wbf + (unsigned)(m0 * Kpad + k0));
            u32x4 g0;
            g0[0] = 1u;                                         // count=1
            g0[1] = lds_as;                                     // lds_addr
            g0[2] = (unsigned)ga;                               // global_addr lo
            g0[3] = (unsigned)((ga >> 32) & 0x01FFFFFFull)      // global_addr hi
                  | (2u << 30);                                 // type=2 (image)
            i32x8 g1;
            g1[0] = (1 << 16)      // data_size = 2 bytes
                  | (1 << 20)      // pad_enable
                  | (3 << 22)      // pad_interval: 16 DWORDs (one 32-bf16 row)
                  | (3 << 25);     // pad_amount: 4 DWORDs  -> row stride 40 bf16
            g1[1] = (int)(32u << 16);   // tensor_dim0 = 32 (tile-local view)
            g1[2] = (int)(32u << 16);   // tensor_dim1 = 32
            g1[3] = (int)(32u << 16);   // tile_dim0   = 32
            g1[4] = 32;                 // tile_dim1   = 32 (tile_dim2 = 0)
            g1[5] = Kpad;               // tensor_dim0_stride (elements)
            g1[6] = 0;
            g1[7] = 0;
            i32x4 gz = {0, 0, 0, 0};
            if (tid < 32) {
#if defined(__clang_major__) && __clang_major__ >= 23
                i32x8 gz8 = {};
                __builtin_amdgcn_tensor_load_to_lds(g0, g1, gz, gz, gz8, 0);
#else
                __builtin_amdgcn_tensor_load_to_lds(g0, g1, gz, gz, 0);
#endif
            }
        }

        // ---- stage im2col B tile (overlaps with TDM) ----
        for (int i = tid; i < 128 * 32; i += 256) {
            int pp = i >> 5, kk = i & 31, k = k0 + kk;
            float v = 0.f;
            int p = p0 + pp;
            if (k < Kreal && p < Npix) {
                int n  = p / OHOW;  int s  = p - n * OHOW;     // OHOW: pow2
                int oy = s / OW;    int ox = s - oy * OW;      // OW: pow2
                int ci, ky, kx;
                if (KS == 3) { ci = k / 9; int t = k - ci * 9; ky = t / 3; kx = t - ky * 3; }
                else         { ci = k; ky = 0; kx = 0; }
                int iy = oy * STRIDE - PAD + ky;
                int ix = ox * STRIDE - PAD + kx;
                if ((unsigned)iy < (unsigned)IH && (unsigned)ix < (unsigned)IW)
                    v = (float)in[(n * CinS + COFF + ci) * IHIW + iy * IW + ix];
            }
            Bs[pp][kk] = (bf16)v;
        }

        if (tid < 32) __builtin_amdgcn_s_wait_tensorcnt(0);
        __syncthreads();

        // A fragment (16x32 bf16): lane half kh holds K = kh*16+j (contiguous)
        bf16x8 alo = *(const bf16x8*)(&As[wm * 16 + row][kh * 16]);
        bf16x8 ahi = *(const bf16x8*)(&As[wm * 16 + row][kh * 16 + 8]);
        v16bf a = __builtin_shufflevector(alo, ahi,
                      0, 1, 2, 3, 4, 5, 6, 7, 8, 9, 10, 11, 12, 13, 14, 15);

        // B fragments (32x16 bf16): lane col = N, K = kh*16 + i (contiguous)
        const bf16* bp0 = &Bs[wn * 32 + row][kh * 16];
        const bf16* bp1 = &Bs[wn * 32 + 16 + row][kh * 16];
        bf16x8 q0 = *(const bf16x8*)bp0;
        bf16x8 q1 = *(const bf16x8*)(bp0 + 8);
        bf16x8 q2 = *(const bf16x8*)bp1;
        bf16x8 q3 = *(const bf16x8*)(bp1 + 8);
        v16bf b0 = __builtin_shufflevector(q0, q1,
                      0, 1, 2, 3, 4, 5, 6, 7, 8, 9, 10, 11, 12, 13, 14, 15);
        v16bf b1 = __builtin_shufflevector(q2, q3,
                      0, 1, 2, 3, 4, 5, 6, 7, 8, 9, 10, 11, 12, 13, 14, 15);

        c0 = __builtin_amdgcn_wmma_f32_16x16x32_bf16(false, a, false, b0,
                                                     (short)0, c0, false, false);
        c1 = __builtin_amdgcn_wmma_f32_16x16x32_bf16(false, a, false, b1,
                                                     (short)0, c1, false, false);
        __syncthreads();
    }

    // Store D (16x16 f32 layout: VGPR r -> M = kh*8 + r, col = lane&15)
    for (int j = 0; j < 2; ++j) {
        v8f cc = j ? c1 : c0;
        int p = p0 + wn * 32 + j * 16 + row;
        if (p < Npix) {
            int n = p / OHOW; int s = p - n * OHOW;
            for (int r = 0; r < 8; ++r) {
                int co = m0 + wm * 16 + kh * 8 + r;
                if (co < Cout) y[(n * Cout + co) * OHOW + s] = cc[r];
            }
        }
    }
}

// --------------------------- BatchNorm pipeline ----------------------------

__launch_bounds__(256)
__global__ void chan_stats(const float* __restrict__ y, float* __restrict__ stats,
                           int C, int SP) {
    __shared__ float s1[256], s2[256];
    int nc = blockIdx.x;            // n*C + c  (one spatial plane per block)
    int c  = nc % C;
    const float* p = y + (long)nc * SP;
    float a = 0.f, b = 0.f;
    for (int i = threadIdx.x; i < SP; i += 256) { float v = p[i]; a += v; b += v * v; }
    s1[threadIdx.x] = a; s2[threadIdx.x] = b;
    __syncthreads();
    for (int st = 128; st > 0; st >>= 1) {
        if (threadIdx.x < st) {
            s1[threadIdx.x] += s1[threadIdx.x + st];
            s2[threadIdx.x] += s2[threadIdx.x + st];
        }
        __syncthreads();
    }
    if (threadIdx.x == 0) {
        atomicAdd(&stats[c], s1[0]);
        atomicAdd(&stats[256 + c], s2[0]);
    }
}

__global__ void bn_finalize(float* __restrict__ stats, const float* __restrict__ g,
                            const float* __restrict__ b, int C, float invCnt) {
    int c = blockIdx.x * 64 + threadIdx.x;
    if (c < C) {
        float m  = stats[c] * invCnt;
        float v  = stats[256 + c] * invCnt - m * m;   // biased var (torch BN)
        float sc = g[c] * rsqrtf(v + 1e-5f);
        stats[512 + c] = sc;
        stats[768 + c] = b[c] - m * sc;
    }
}

__launch_bounds__(256)
__global__ void bn_apply(const float* __restrict__ y, const float* __restrict__ stats,
                         float* __restrict__ o32, bf16* __restrict__ o16,
                         int n, int C, int SP) {
    int i = blockIdx.x * 256 + threadIdx.x;
    if (i >= n) return;
    int c = (i / SP) % C;
    float v = fmaf(y[i], stats[512 + c], stats[768 + c]);
    v = fmaxf(v, 0.f);
    if (o32) o32[i] = v;
    if (o16) o16[i] = (bf16)v;
}

// ------------------- per-sample transposed conv + losses -------------------

__launch_bounds__(256)
__global__ void deconv_k(const float* __restrict__ A, const float* __restrict__ S,
                         float* __restrict__ recon) {
    __shared__ float Ssl[64][65];
    int nc = blockIdx.x; int n = nc / 3; int c = nc % 3;
    for (int i = threadIdx.x; i < 64 * 64; i += 256) {
        int d = i >> 6; int t = i & 63;
        Ssl[d][t] = S[((n * 64 + d) * 3 + c) * 64 + t];
    }
    __syncthreads();
    const float* An = A + (long)n * 64 * 256;
    for (int q = 0; q < 4; ++q) {
        int pix = threadIdx.x + q * 256;       // 0..1023 -> (oy,ox)
        int oy = pix >> 5, ox = pix & 31;
        float acc = 0.f;
        for (int iy = 0; iy < 16; ++iy) {
            int ky = oy + 3 - 2 * iy;
            if ((unsigned)ky >= 8u) continue;
            for (int ix = 0; ix < 16; ++ix) {
                int kx = ox + 3 - 2 * ix;
                if ((unsigned)kx >= 8u) continue;
                const float* Ap = An + iy * 16 + ix;
                int sidx = ky * 8 + kx;
                float sum = 0.f;
                for (int d = 0; d < 64; ++d)
                    sum = fmaf(Ap[d * 256], Ssl[d][sidx], sum);
                acc += sum;
            }
        }
        recon[(n * 3 + c) * 1024 + pix] = acc;
    }
}

__launch_bounds__(256)
__global__ void loss1_k(const float* __restrict__ x, const float* __restrict__ recon,
                        float* __restrict__ out, int n) {
    __shared__ float red[256];
    int i = blockIdx.x * 256 + threadIdx.x;
    float v = 0.f;
    if (i < n) {
        float a = 1.f / (1.f + __expf(-x[i]));
        float b = 1.f / (1.f + __expf(-recon[i]));
        float d = a - b;
        v = d * d;
    }
    red[threadIdx.x] = v; __syncthreads();
    for (int st = 128; st > 0; st >>= 1) {
        if (threadIdx.x < st) red[threadIdx.x] += red[threadIdx.x + st];
        __syncthreads();
    }
    if (threadIdx.x == 0) atomicAdd(out, red[0] * (0.1f / (float)n));
}

__launch_bounds__(256)
__global__ void loss_s_k(const float* __restrict__ S, float* __restrict__ l2,
                         float* __restrict__ l3) {
    __shared__ float r2[256], r3[256];
    int g = blockIdx.x * 256 + threadIdx.x;
    float s2 = 0.f, s3 = 0.f;
    const int NG = 1024 * 192;
    if (g < NG) {
        int n = g / 192; int r = g - n * 192;
        const float* base = S + (long)n * 64 * 192 + r;
        float mx = -1e30f;
        for (int d = 0; d < 64; ++d) {
            float v = base[d * 192];
            mx = fmaxf(mx, v);
            s2 += fmaxf(v, 0.f);
        }
        float se = 0.f;
        for (int d = 0; d < 64; ++d) se += __expf(base[d * 192] - mx);
        float lse = __logf(se) + mx;
        for (int d = 0; d < 64; ++d) {
            float lp = base[d * 192] - lse;
            s3 += __expf(lp) * lp;
        }
    }
    r2[threadIdx.x] = s2; r3[threadIdx.x] = s3; __syncthreads();
    for (int st = 128; st > 0; st >>= 1) {
        if (threadIdx.x < st) {
            r2[threadIdx.x] += r2[threadIdx.x + st];
            r3[threadIdx.x] += r3[threadIdx.x + st];
        }
        __syncthreads();
    }
    if (threadIdx.x == 0) {
        const float inv = 1.0f / 12582912.0f;
        atomicAdd(l2, r2[0] * (0.1f * inv));
        atomicAdd(l3, r3[0] * (-0.1f * inv));
    }
}

// ------------------------------- orchestration -----------------------------

template<int Cin, int CinS, int COFF, int Cout,
         int IH, int IW, int OH, int OW, int STRIDE, int PAD, int KS>
static void run_layer(const bf16* in, const bf16* wbf, const float* g, const float* b,
                      float* RAW, float* ST, float* o32, bf16* o16, hipStream_t stream) {
    constexpr int Npix = 1024 * OH * OW;
    dim3 grid(Npix / 128, Cout / 32);
    conv_wmma<Cin, CinS, COFF, Cout, IH, IW, OH, OW, STRIDE, PAD, KS>
        <<<grid, 256, 0, stream>>>(in, wbf, RAW);
    zero_f32<<<2, 256, 0, stream>>>(ST, 512);
    chan_stats<<<1024 * Cout, 256, 0, stream>>>(RAW, ST, Cout, OH * OW);
    bn_finalize<<<(Cout + 63) / 64, 64, 0, stream>>>(ST, g, b, Cout,
                                                     1.0f / (1024.f * OH * OW));
    constexpr int n = Npix * Cout;
    bn_apply<<<(n + 255) / 256, 256, 0, stream>>>(RAW, ST, o32, o16, n, Cout, OH * OW);
}

extern "C" void kernel_launch(void* const* d_in, const int* in_sizes, int n_in,
                              void* d_out, int out_size, void* d_ws, size_t ws_size,
                              hipStream_t stream) {
    const float* x   = (const float*)d_in[0];
    const float* w1  = (const float*)d_in[1];
    const float* w2  = (const float*)d_in[2];
    const float* w3  = (const float*)d_in[3];
    const float* wa  = (const float*)d_in[4];
    const float* ws1 = (const float*)d_in[5];
    const float* ws2 = (const float*)d_in[6];
    const float* gb[12];
    for (int i = 0; i < 12; ++i) gb[i] = (const float*)d_in[7 + i];

    char*  ws   = (char*)d_ws;
    float* RAW  = (float*)ws;                                  // 268,435,456 B
    bf16*  BUF0 = (bf16*)(ws + 268435456);                     // 134,217,728 B
    bf16*  BUF1 = (bf16*)(ws + 268435456 + 134217728);         // 134,217,728 B
    bf16*  XB   = (bf16*)(ws + 268435456 + 2 * 134217728);     //   6,291,456 B
    float* ST   = (float*)(ws + 268435456 + 2 * 134217728 + 8388608);
    bf16*  WB   = (bf16*)(ws + 268435456 + 2 * 134217728 + 8388608 + 8192);
    // packed weight offsets (elements, Kpad-padded rows)
    bf16* wb1  = WB;            // 32  x 32   = 1024
    bf16* wb2  = WB + 1024;     // 64  x 288  = 18432
    bf16* wb3  = WB + 19456;    // 64  x 576  = 36864
    bf16* wbA  = WB + 56320;    // 64  x 32   = 2048
    bf16* wbS1 = WB + 58368;    // 192 x 288  = 55296
    bf16* wbS2 = WB + 113664;   // 192 x 1728 = 331776

    float* out   = (float*)d_out;
    float* A_out = out;                     // 1024*64*16*16 = 16,777,216
    float* S_out = out + 16777216;          // 1024*192*8*8  = 12,582,912
    float* Lp    = out + 29360128;          // l1, l2, l3 scalars

    zero_f32<<<1, 256, 0, stream>>>(Lp, 3);

    const int nx = 1024 * 3 * 32 * 32;
    pack_bf16_k<<<(nx + 255) / 256, 256, 0, stream>>>(x, XB, nx);

    wpack_k<<<(1024   + 255) / 256, 256, 0, stream>>>(w1,  wb1,   32,   27,   32);
    wpack_k<<<(18432  + 255) / 256, 256, 0, stream>>>(w2,  wb2,   64,  288,  288);
    wpack_k<<<(36864  + 255) / 256, 256, 0, stream>>>(w3,  wb3,   64,  576,  576);
    wpack_k<<<(2048   + 255) / 256, 256, 0, stream>>>(wa,  wbA,   64,   32,   32);
    wpack_k<<<(55296  + 255) / 256, 256, 0, stream>>>(ws1, wbS1, 192,  288,  288);
    wpack_k<<<(331776 + 255) / 256, 256, 0, stream>>>(ws2, wbS2, 192, 1728, 1728);

    //         Cin CinS OFF Cout  IH  IW  OH  OW  S  P  KS
    run_layer<  3,   3,  0,  32,  32, 32, 32, 32, 1, 1, 3>(XB,   wb1,  gb[0], gb[1],  RAW, ST, nullptr, BUF0, stream);
    run_layer< 32,  32,  0,  64,  32, 32, 32, 32, 1, 1, 3>(BUF0, wb2,  gb[2], gb[3],  RAW, ST, nullptr, BUF1, stream);
    run_layer< 64,  64,  0,  64,  32, 32, 16, 16, 2, 1, 3>(BUF1, wb3,  gb[4], gb[5],  RAW, ST, nullptr, BUF0, stream);
    run_layer< 32,  64,  0,  64,  16, 16, 16, 16, 1, 0, 1>(BUF0, wbA,  gb[6], gb[7],  RAW, ST, A_out,  nullptr, stream);
    run_layer< 32,  64, 32, 192,  16, 16, 16, 16, 1, 1, 3>(BUF0, wbS1, gb[8], gb[9],  RAW, ST, nullptr, BUF1, stream);
    run_layer<192, 192,  0, 192,  16, 16,  8,  8, 2, 1, 3>(BUF1, wbS2, gb[10],gb[11], RAW, ST, S_out,  nullptr, stream);

    float* RECON = RAW;   // RAW free now; 12.6 MB
    deconv_k<<<1024 * 3, 256, 0, stream>>>(A_out, S_out, RECON);
    loss1_k<<<(nx + 255) / 256, 256, 0, stream>>>(x, RECON, Lp, nx);
    loss_s_k<<<(1024 * 192 + 255) / 256, 256, 0, stream>>>(S_out, Lp + 1, Lp + 2);
}
